// sMoE_6631429505580
// MI455X (gfx1250) — matrix operand order repaired and checked by reading, not compiled, not run on hardware
//
#include <hip/hip_runtime.h>
#include <hip/hip_bf16.h>
#include <math.h>

// ---------------------------------------------------------------------------
// MoE (top-2 of 8 experts), MI455X / gfx1250, wave32, bf16 WMMA 16x16x32.
// Sparse-routed: only the B*K = 16384 selected (row, expert) pairs computed.
// ---------------------------------------------------------------------------

#if defined(__has_builtin)
#if __has_builtin(__builtin_amdgcn_sched_barrier)
#define SCHED_FENCE() __builtin_amdgcn_sched_barrier(0)
#endif
#endif
#ifndef SCHED_FENCE
#define SCHED_FENCE() ((void)0)
#endif

typedef __attribute__((ext_vector_type(8)))  float  v8f;
typedef __attribute__((ext_vector_type(16))) __bf16 v16bf;
typedef __attribute__((ext_vector_type(8)))  __bf16 bf16x8;

union AFrag { struct { bf16x8 lo, hi; } p; v16bf v; };

constexpr int Bn = 8192;   // tokens
constexpr int Dd = 1024;   // model dim
constexpr int Hh = 4096;   // hidden dim
constexpr int Oo = 1024;   // output dim
constexpr int Ee = 8;      // experts
constexpr int TM = 128;    // M tile (rows per block)
constexpr int TN = 128;    // N tile (cols per block, 8 frags x 16)
constexpr int TK = 32;     // K step (bf16 wmma depth)
constexpr int LDSK = TK + 8; // 40 halfs = 80B stride: 16B aligned + conflict free

__device__ __forceinline__ float wave_sum(float v) {
  #pragma unroll
  for (int m = 16; m; m >>= 1) v += __shfl_xor(v, m, 32);
  return v;
}
__device__ __forceinline__ float wave_max(float v) {
  #pragma unroll
  for (int m = 16; m; m >>= 1) v = fmaxf(v, __shfl_xor(v, m, 32));
  return v;
}

// ---------------------------------------------------------------- init ------
__global__ void moe_init(int* __restrict__ counts) {
  if (threadIdx.x < Ee) counts[threadIdx.x] = 0;
}

// -------------------------------------------------------------- gating ------
__global__ __launch_bounds__(256) void moe_gating(
    const float* __restrict__ x, const float* __restrict__ wg,
    int* __restrict__ topE, float* __restrict__ topG, int* __restrict__ topPos,
    int* __restrict__ counts, int* __restrict__ rowlist)
{
  const int wave = threadIdx.x >> 5, lane = threadIdx.x & 31;
  const int row  = blockIdx.x * 8 + wave;
  float acc[Ee];
  #pragma unroll
  for (int e = 0; e < Ee; ++e) acc[e] = 0.f;
  const float* xr = x + (size_t)row * Dd;
  for (int d = lane; d < Dd; d += 32) {
    const float xv = xr[d];
    const float* w = wg + (size_t)d * Ee;
    #pragma unroll
    for (int e = 0; e < Ee; ++e) acc[e] = fmaf(xv, w[e], acc[e]);
  }
  #pragma unroll
  for (int e = 0; e < Ee; ++e) acc[e] = wave_sum(acc[e]);
  if (lane == 0) {
    int i1 = 0; float v1 = acc[0];
    #pragma unroll
    for (int e = 1; e < Ee; ++e) if (acc[e] > v1) { v1 = acc[e]; i1 = e; }
    int i2 = (i1 == 0) ? 1 : 0; float v2 = acc[i2];
    #pragma unroll
    for (int e = 0; e < Ee; ++e) if (e != i1 && acc[e] > v2) { v2 = acc[e]; i2 = e; }
    const float ex  = __expf(v2 - v1);        // v2 <= v1, stable
    const float inv = 1.f / (1.f + ex);
    const float g1 = inv, g2 = ex * inv;
    const int p1 = atomicAdd(&counts[i1], 1);
    const int p2 = atomicAdd(&counts[i2], 1);
    rowlist[i1 * Bn + p1] = row;
    rowlist[i2 * Bn + p2] = row;
    topE[row * 2 + 0] = i1;  topE[row * 2 + 1] = i2;
    topG[row * 2 + 0] = g1;  topG[row * 2 + 1] = g2;
    topPos[row * 2 + 0] = p1; topPos[row * 2 + 1] = p2;
  }
}

// ------------------------------------------------------------- offsets ------
__global__ void moe_offsets(const int* __restrict__ counts, int* __restrict__ offsets) {
  if (threadIdx.x == 0) {
    int s = 0;
    for (int e = 0; e < Ee; ++e) { offsets[e] = s; s += counts[e]; }
    offsets[Ee] = s;
  }
}

// ---------------------------------------------------------------- loss ------
__global__ __launch_bounds__(256) void moe_loss(
    const int* __restrict__ topE, const float* __restrict__ topG,
    const int* __restrict__ counts, float* __restrict__ out_loss)
{
  __shared__ float red[256];
  const int tid = threadIdx.x;
  float imp[Ee];
  #pragma unroll
  for (int e = 0; e < Ee; ++e) imp[e] = 0.f;
  for (int r = tid; r < Bn; r += 256) {
    imp[topE[r * 2 + 0]] += topG[r * 2 + 0];
    imp[topE[r * 2 + 1]] += topG[r * 2 + 1];
  }
  float impTot[Ee];
  for (int e = 0; e < Ee; ++e) {
    red[tid] = imp[e];
    __syncthreads();
    for (int s = 128; s; s >>= 1) {
      if (tid < s) red[tid] += red[tid + s];
      __syncthreads();
    }
    impTot[e] = red[0];
    __syncthreads();
  }
  if (tid == 0) {
    float mi = 0.f, ml = 0.f;
    for (int e = 0; e < Ee; ++e) { mi += impTot[e]; ml += (float)counts[e]; }
    mi /= Ee; ml /= Ee;
    float vi = 0.f, vl = 0.f;
    for (int e = 0; e < Ee; ++e) {
      const float di = impTot[e] - mi;        vi += di * di;
      const float dl = (float)counts[e] - ml; vl += dl * dl;
    }
    vi /= (Ee - 1); vl /= (Ee - 1);            // ddof=1 (torch.var)
    const float cv_i = vi / (mi * mi + 1e-10f);
    const float cv_l = vl / (ml * ml + 1e-10f);
    out_loss[0] = (cv_i + cv_l) * 1e-2f;
  }
}

// ---------------------------------------------------------------- FFN1 ------
// H = relu(gather(x) @ W1[e] + b1[e]) -> Hbuf (bf16, compact slot space).
// Double-buffered LDS, one barrier per K-step; all 8 B-fragments preloaded
// behind a scheduling fence so the 8 WMMAs issue back-to-back.
__global__ __launch_bounds__(256) void moe_ffn1(
    const float* __restrict__ x, const float* __restrict__ W1, const float* __restrict__ b1,
    const int* __restrict__ counts, const int* __restrict__ offsets,
    const int* __restrict__ rowlist, __bf16* __restrict__ Hbuf)
{
  const int e   = blockIdx.z;
  const int cnt = counts[e];
  const int mt  = blockIdx.x;
  if (mt * TM >= cnt) return;
  const int n0  = blockIdx.y * TN;

  __shared__ int    rows_s[TM];
  __shared__ __bf16 Asub[2][TM][LDSK];
  __shared__ __bf16 Bsub[2][TN][LDSK];

  const int tid = threadIdx.x, wave = tid >> 5, lane = tid & 31;
  if (tid < TM) {
    int ig = mt * TM + tid;
    if (ig >= cnt) ig = cnt - 1;               // clamp: padded rows recompute a
    rows_s[tid] = rowlist[e * Bn + ig];        // valid row; stores are guarded
  }
  __syncthreads();

  // A staging role: thread -> (row ar, 16-wide k chunk at akb)
  const int ar  = tid >> 1;
  const int akb = (tid & 1) * 16;
  const float* asrc = x + (size_t)rows_s[ar] * Dd + akb;
  // B staging role: thread -> (col bc, 16-wide k chunk at bkg), coalesced over bc
  const int bc  = tid & 127;
  const int bkg = (tid >> 7) * 16;
  const float* Wcol = W1 + (size_t)e * Dd * Hh + n0 + bc;

  auto stage = [&](int kk, bf16x8& a0, bf16x8& a1, bf16x8& b0v, bf16x8& b1v) {
    const float* as = asrc + kk;
    #pragma unroll
    for (int i = 0; i < 8; ++i) a0[i] = (__bf16)as[i];
    #pragma unroll
    for (int i = 0; i < 8; ++i) a1[i] = (__bf16)as[8 + i];
    const float* bs = Wcol + (size_t)(kk + bkg) * Hh;
    if (kk + TK < Dd) __builtin_prefetch(bs + (size_t)TK * Hh, 0, 1);
    float bvals[16];
    #pragma unroll
    for (int i = 0; i < 16; ++i) bvals[i] = bs[(size_t)i * Hh];
    #pragma unroll
    for (int i = 0; i < 8; ++i) { b0v[i] = (__bf16)bvals[i]; b1v[i] = (__bf16)bvals[8 + i]; }
  };

  v8f acc[8];
  #pragma unroll
  for (int nf = 0; nf < 8; ++nf)
    #pragma unroll
    for (int i = 0; i < 8; ++i) acc[nf][i] = 0.f;

  const int lm = lane & 15, half = lane >> 4;

  // prologue: stage k=0 into buffer 0
  {
    bf16x8 a0, a1, b0v, b1v;
    stage(0, a0, a1, b0v, b1v);
    *(bf16x8*)&Asub[0][ar][akb]     = a0;
    *(bf16x8*)&Asub[0][ar][akb + 8] = a1;
    *(bf16x8*)&Bsub[0][bc][bkg]     = b0v;
    *(bf16x8*)&Bsub[0][bc][bkg + 8] = b1v;
  }
  __syncthreads();

  int p = 0;
  for (int k0 = 0; k0 < Dd; k0 += TK) {
    const bool has_next = (k0 + TK) < Dd;
    // issue next step's global loads + converts early
    bf16x8 na0, na1, nb0, nb1;
    if (has_next) stage(k0 + TK, na0, na1, nb0, nb1);

    // compute from buffer p: preload A frag + ALL 8 B frags, then fence,
    // then 8 independent WMMAs (partial dscnt waits, back-to-back issue).
    AFrag af;
    af.p.lo = *(const bf16x8*)&Asub[p][wave * 16 + lm][half * 8];
    af.p.hi = *(const bf16x8*)&Asub[p][wave * 16 + lm][half * 8 + 16];
    AFrag bfr[8];
    #pragma unroll
    for (int nf = 0; nf < 8; ++nf) {
      bfr[nf].p.lo = *(const bf16x8*)&Bsub[p][nf * 16 + lm][half * 8];
      bfr[nf].p.hi = *(const bf16x8*)&Bsub[p][nf * 16 + lm][half * 8 + 16];
    }
    SCHED_FENCE();
    #pragma unroll
    for (int nf = 0; nf < 8; ++nf)
      acc[nf] = __builtin_amdgcn_wmma_f32_16x16x32_bf16(
          false, af.v, false, bfr[nf].v, (short)0, acc[nf], false, false);
    SCHED_FENCE();

    // publish next buffer; single barrier retires reads of p and stores to p^1
    if (has_next) {
      *(bf16x8*)&Asub[p ^ 1][ar][akb]     = na0;
      *(bf16x8*)&Asub[p ^ 1][ar][akb + 8] = na1;
      *(bf16x8*)&Bsub[p ^ 1][bc][bkg]     = nb0;
      *(bf16x8*)&Bsub[p ^ 1][bc][bkg + 8] = nb1;
      __syncthreads();
    }
    p ^= 1;
  }

  // ---- epilogue: +bias, relu, bf16 store into compact Hbuf ----
  const int slot_base = offsets[e] + mt * TM + wave * 16;
  #pragma unroll
  for (int nf = 0; nf < 8; ++nf) {
    const int h = n0 + nf * 16 + lm;
    const float bias = b1[e * Hh + h];
    #pragma unroll
    for (int i = 0; i < 8; ++i) {
      const int m  = i + half * 8;
      const int ig = mt * TM + wave * 16 + m;
      if (ig < cnt) {
        float v = acc[nf][i] + bias;
        v = v > 0.f ? v : 0.f;
        Hbuf[(size_t)(slot_base + m) * Hh + h] = (__bf16)v;
      }
    }
  }
}

// ---------------------------------------------------------------- FFN2 ------
// Out2 = Hbuf @ W2[e] + b2[e] (fp32 logits, pre-softmax), K over H=4096.
__global__ __launch_bounds__(256) void moe_ffn2(
    const __bf16* __restrict__ Hbuf, const float* __restrict__ W2, const float* __restrict__ b2,
    const int* __restrict__ counts, const int* __restrict__ offsets,
    float* __restrict__ Out2)
{
  const int e   = blockIdx.z;
  const int cnt = counts[e];
  const int mt  = blockIdx.x;
  if (mt * TM >= cnt) return;
  const int n0   = blockIdx.y * TN;
  const int base = offsets[e];

  __shared__ __bf16 Asub[2][TM][LDSK];
  __shared__ __bf16 Bsub[2][TN][LDSK];

  const int tid = threadIdx.x, wave = tid >> 5, lane = tid & 31;
  const int ar  = tid >> 1;
  const int akb = (tid & 1) * 16;
  int ars = mt * TM + ar;
  if (ars >= cnt) ars = cnt - 1;               // clamp: padded rows duplicate
  const __bf16* asrc = Hbuf + (size_t)(base + ars) * Hh + akb;
  const int bc  = tid & 127;
  const int bkg = (tid >> 7) * 16;
  const float* Wcol = W2 + (size_t)e * Hh * Oo + n0 + bc;

  auto stage = [&](int kk, bf16x8& a0, bf16x8& a1, bf16x8& b0v, bf16x8& b1v) {
    const __bf16* as = asrc + kk;
    a0 = *(const bf16x8*)(as);
    a1 = *(const bf16x8*)(as + 8);
    const float* bs = Wcol + (size_t)(kk + bkg) * Oo;
    if (kk + TK < Hh) __builtin_prefetch(bs + (size_t)TK * Oo, 0, 1);
    float bvals[16];
    #pragma unroll
    for (int i = 0; i < 16; ++i) bvals[i] = bs[(size_t)i * Oo];
    #pragma unroll
    for (int i = 0; i < 8; ++i) { b0v[i] = (__bf16)bvals[i]; b1v[i] = (__bf16)bvals[8 + i]; }
  };

  v8f acc[8];
  #pragma unroll
  for (int nf = 0; nf < 8; ++nf)
    #pragma unroll
    for (int i = 0; i < 8; ++i) acc[nf][i] = 0.f;

  const int lm = lane & 15, half = lane >> 4;

  {
    bf16x8 a0, a1, b0v, b1v;
    stage(0, a0, a1, b0v, b1v);
    *(bf16x8*)&Asub[0][ar][akb]     = a0;
    *(bf16x8*)&Asub[0][ar][akb + 8] = a1;
    *(bf16x8*)&Bsub[0][bc][bkg]     = b0v;
    *(bf16x8*)&Bsub[0][bc][bkg + 8] = b1v;
  }
  __syncthreads();

  int p = 0;
  for (int k0 = 0; k0 < Hh; k0 += TK) {
    const bool has_next = (k0 + TK) < Hh;
    bf16x8 na0, na1, nb0, nb1;
    if (has_next) stage(k0 + TK, na0, na1, nb0, nb1);

    AFrag af;
    af.p.lo = *(const bf16x8*)&Asub[p][wave * 16 + lm][half * 8];
    af.p.hi = *(const bf16x8*)&Asub[p][wave * 16 + lm][half * 8 + 16];
    AFrag bfr[8];
    #pragma unroll
    for (int nf = 0; nf < 8; ++nf) {
      bfr[nf].p.lo = *(const bf16x8*)&Bsub[p][nf * 16 + lm][half * 8];
      bfr[nf].p.hi = *(const bf16x8*)&Bsub[p][nf * 16 + lm][half * 8 + 16];
    }
    SCHED_FENCE();
    #pragma unroll
    for (int nf = 0; nf < 8; ++nf)
      acc[nf] = __builtin_amdgcn_wmma_f32_16x16x32_bf16(
          false, af.v, false, bfr[nf].v, (short)0, acc[nf], false, false);
    SCHED_FENCE();

    if (has_next) {
      *(bf16x8*)&Asub[p ^ 1][ar][akb]     = na0;
      *(bf16x8*)&Asub[p ^ 1][ar][akb + 8] = na1;
      *(bf16x8*)&Bsub[p ^ 1][bc][bkg]     = nb0;
      *(bf16x8*)&Bsub[p ^ 1][bc][bkg + 8] = nb1;
      __syncthreads();
    }
    p ^= 1;
  }

  const int slot_base = base + mt * TM + wave * 16;
  #pragma unroll
  for (int nf = 0; nf < 8; ++nf) {
    const int o = n0 + nf * 16 + lm;
    const float bias = b2[e * Oo + o];
    #pragma unroll
    for (int i = 0; i < 8; ++i) {
      const int m  = i + half * 8;
      const int ig = mt * TM + wave * 16 + m;
      if (ig < cnt) Out2[(size_t)(slot_base + m) * Oo + o] = acc[nf][i] + bias;
    }
  }
}

// ------------------------------------------------------------- combine ------
__global__ __launch_bounds__(256) void moe_combine(
    const float* __restrict__ Out2, const int* __restrict__ topE,
    const float* __restrict__ topG, const int* __restrict__ topPos,
    const int* __restrict__ offsets, float* __restrict__ y)
{
  const int wave = threadIdx.x >> 5, lane = threadIdx.x & 31;
  const int row  = blockIdx.x * 8 + wave;
  float out[32];
  #pragma unroll
  for (int i = 0; i < 32; ++i) out[i] = 0.f;

  #pragma unroll
  for (int j = 0; j < 2; ++j) {
    const int   e    = topE[row * 2 + j];
    const float g    = topG[row * 2 + j];
    const int   slot = offsets[e] + topPos[row * 2 + j];
    const float* src = Out2 + (size_t)slot * Oo;
    float vals[32];
    float m = -INFINITY;
    #pragma unroll
    for (int i = 0; i < 32; ++i) { vals[i] = src[lane + 32 * i]; m = fmaxf(m, vals[i]); }
    m = wave_max(m);
    float s = 0.f;
    #pragma unroll
    for (int i = 0; i < 32; ++i) { vals[i] = __expf(vals[i] - m); s += vals[i]; }
    s = wave_sum(s);
    const float sc = g / s;
    #pragma unroll
    for (int i = 0; i < 32; ++i) out[i] = fmaf(sc, vals[i], out[i]);
  }
  float* dst = y + (size_t)row * Oo;
  #pragma unroll
  for (int i = 0; i < 32; ++i) dst[lane + 32 * i] = out[i];
}

// ---------------------------------------------------------------------------
extern "C" void kernel_launch(void* const* d_in, const int* in_sizes, int n_in,
                              void* d_out, int out_size, void* d_ws, size_t ws_size,
                              hipStream_t stream)
{
  (void)in_sizes; (void)n_in; (void)out_size; (void)ws_size;
  const float* x  = (const float*)d_in[0];
  const float* wg = (const float*)d_in[1];
  const float* W1 = (const float*)d_in[2];
  const float* b1 = (const float*)d_in[3];
  const float* W2 = (const float*)d_in[4];
  const float* b2 = (const float*)d_in[5];
  float* y = (float*)d_out;                   // [B*O] then loss at [B*O]

  char* ws = (char*)d_ws;
  size_t off = 0;
  auto carve = [&](size_t bytes) -> void* {
    void* p = ws + off;
    off = (off + bytes + 255) & ~(size_t)255;
    return p;
  };
  int*    counts  = (int*)   carve(Ee * sizeof(int));
  int*    offsets = (int*)   carve((Ee + 1) * sizeof(int));
  int*    topE    = (int*)   carve((size_t)Bn * 2 * sizeof(int));
  float*  topG    = (float*) carve((size_t)Bn * 2 * sizeof(float));
  int*    topPos  = (int*)   carve((size_t)Bn * 2 * sizeof(int));
  int*    rowlist = (int*)   carve((size_t)Ee * Bn * sizeof(int));
  __bf16* Hbuf    = (__bf16*)carve((size_t)Bn * 2 * Hh * sizeof(__bf16));  // 128 MB
  float*  Out2    = (float*) carve((size_t)Bn * 2 * Oo * sizeof(float));   //  64 MB

  moe_init<<<1, 32, 0, stream>>>(counts);
  moe_gating<<<Bn / 8, 256, 0, stream>>>(x, wg, topE, topG, topPos, counts, rowlist);
  moe_offsets<<<1, 32, 0, stream>>>(counts, offsets);
  moe_loss<<<1, 256, 0, stream>>>(topE, topG, counts, y + (size_t)Bn * Oo);
  moe_ffn1<<<dim3(Bn / TM, Hh / TN, Ee), 256, 0, stream>>>(
      x, W1, b1, counts, offsets, rowlist, Hbuf);
  moe_ffn2<<<dim3(Bn / TM, Oo / TN, Ee), 256, 0, stream>>>(
      Hbuf, W2, b2, counts, offsets, Out2);
  moe_combine<<<Bn / 8, 256, 0, stream>>>(Out2, topE, topG, topPos, offsets, y);
}
